// ReaxFF_nn_force_46488726012371
// MI455X (gfx1250) — compile-verified, matrix operand order checked by reading, not run on hardware
//
#include <hip/hip_runtime.h>
#include <hip/hip_bf16.h>
#include <math.h>

typedef __attribute__((ext_vector_type(16))) _Float16 v16h;
typedef __attribute__((ext_vector_type(8)))  float    v8f;

#define BB  48
#define NN  512
#define HH  8

__device__ __forceinline__ float taperf(float r, float rmin, float rmax) {
  float r3  = (r > rmax) ? 1.0f : 0.0f;
  bool  ok  = (r <= rmax) && (r > rmin);
  float r2  = ok ? r : 0.0f;
  float r20 = ok ? 1.0f : 0.0f;
  float d   = rmin - rmax;
  float rterm = 1.0f / (d * d * d);
  float rm  = rmin * r20;
  float rd  = rm - r2;
  float trm1 = rm + 2.0f * r2 - 3.0f * rmax * r20;
  return rterm * rd * rd * trm1 + r3;
}

// fast sigmoid: v_exp_f32 + v_rcp_f32 (no IEEE divide sequence)
__device__ __forceinline__ float sigf(float x) {
  return __builtin_amdgcn_rcpf(1.0f + __expf(-x));
}

__device__ __forceinline__ v8f wmma16(v16h a, v16h b, v8f c) {
  // D = A(16x32,f16) x B(32x16,f16) + C(16x16,f32)
  return __builtin_amdgcn_wmma_f32_16x16x32_f16(false, a, false, b, (short)0, c,
                                                false, false);
}

// Wave-local LDS ordering: DS ops from one wave are processed in order at the
// LDS unit (ISA 01_flow_sync §7.3); we only need a compiler fence + dscnt wait.
__device__ __forceinline__ void lds_fence() {
  asm volatile("s_wait_dscnt 0" ::: "memory");
}

// ---------------------------------------------------------------- zero init
__global__ void zero_kernel(float* __restrict__ out, float* __restrict__ ws, int nws) {
  int idx = blockIdx.x * blockDim.x + threadIdx.x;
  if (idx < BB)  out[idx] = 0.0f;
  if (idx < nws) ws[idx]  = 0.0f;
}

// ---------------------------------------------------------------- bonds + MLP (WMMA)
// grid = 1536 blocks x 256 threads; 8 waves/block; 1 wave = 16 bond rows.
// Tiles: seg0 4800, seg1 4800, seg2 2688 (total 12288 = 1536*8).
__global__ __launch_bounds__(256)
void bond_kernel(const float* __restrict__ x, const float* __restrict__ cell,
                 const float* __restrict__ rcell, const int* __restrict__ bdid,
                 const float* __restrict__ rosi, const float* __restrict__ pbo1,
                 const float* __restrict__ pbo2, const float* __restrict__ ropi,
                 const float* __restrict__ pbo3, const float* __restrict__ pbo4,
                 const float* __restrict__ ropp, const float* __restrict__ pbo5,
                 const float* __restrict__ pbo6, const float* __restrict__ Desi,
                 const float* __restrict__ fwi, const float* __restrict__ fw,
                 const float* __restrict__ fb,  const float* __restrict__ fwo,
                 const float* __restrict__ fbo,
                 float* __restrict__ Delta, float* __restrict__ Dpi,
                 float* __restrict__ SOa,   float* __restrict__ out) {
  __shared__ _Float16 lds[8][16 * 16];  // per-wave 16x16 f16 staging tile

  const int wave = threadIdx.x >> 5;
  const int lane = threadIdx.x & 31;
  const int tile = blockIdx.x * 8 + wave;

  int t, lt, s0, tpb;
  if (tile < 4800)      { t = 0; lt = tile;        s0 = 0;    tpb = 100; }
  else if (tile < 9600) { t = 1; lt = tile - 4800; s0 = 1600; tpb = 100; }
  else                  { t = 2; lt = tile - 9600; s0 = 3200; tpb = 56;  }
  const int b  = lt / tpb;      // all 16 rows of a tile share one batch
  const int kt = lt % tpb;

  const int col = lane & 15;
  const bool lo = lane < 16;

  // ---- per-segment weight fragments (B layout: lane = column N, elem = K)
  v16h bw1 = {}, bw2 = {}, bw3 = {}, bw4 = {};
  if (lo && col < HH) {
    for (int e = 0; e < 3; ++e)
      bw1[e] = (_Float16)fwi[(t * 3 + e) * HH + col];
    for (int e = 0; e < HH; ++e) {
      bw2[e] = (_Float16)fw[(((t * 2 + 0) * HH) + e) * HH + col];
      bw3[e] = (_Float16)fw[(((t * 2 + 1) * HH) + e) * HH + col];
    }
  }
  if (lo && col == 0)
    for (int e = 0; e < HH; ++e) bw4[e] = (_Float16)fwo[t * HH + e];

  // ---- bias accumulators (C layout: bias depends only on column)
  v8f cb2 = {}, cb3 = {}, cb4 = {}, czero = {};
  {
    float b2 = (col < HH) ? fb[(t * 2 + 0) * HH + col] : 0.0f;
    float b3 = (col < HH) ? fb[(t * 2 + 1) * HH + col] : 0.0f;
    float b4 = (col == 0) ? fbo[t] : 0.0f;
    for (int m = 0; m < 8; ++m) { cb2[m] = b2; cb3[m] = b3; cb4[m] = b4; }
  }

  // ---- scalar phase: 16 lanes compute 16 bond rows
  v16h a1 = {};
  if (lo) {
    const int k = s0 + kt * 16 + lane;
    const int i = bdid[2 * k], j = bdid[2 * k + 1];
    const float* xi = x + ((size_t)b * NN + i) * 3;
    const float* xj = x + ((size_t)b * NN + j) * 3;
    float v0 = xi[0] - xj[0], v1 = xi[1] - xj[1], v2 = xi[2] - xj[2];
    float f0 = v0 * rcell[0] + v1 * rcell[3] + v2 * rcell[6];
    float f1 = v0 * rcell[1] + v1 * rcell[4] + v2 * rcell[7];
    float f2 = v0 * rcell[2] + v1 * rcell[5] + v2 * rcell[8];
    f0 = (f0 - 0.5f > 0.0f) ? f0 - 1.0f : f0;  f0 = (f0 + 0.5f < 0.0f) ? f0 + 1.0f : f0;
    f1 = (f1 - 0.5f > 0.0f) ? f1 - 1.0f : f1;  f1 = (f1 + 0.5f < 0.0f) ? f1 + 1.0f : f1;
    f2 = (f2 - 0.5f > 0.0f) ? f2 - 1.0f : f2;  f2 = (f2 + 0.5f < 0.0f) ? f2 + 1.0f : f2;
    float w0 = f0 * cell[0] + f1 * cell[3] + f2 * cell[6];
    float w1 = f0 * cell[1] + f1 * cell[4] + f2 * cell[7];
    float w2 = f0 * cell[2] + f1 * cell[5] + f2 * cell[8];
    float r = sqrtf(w0 * w0 + w1 * w1 + w2 * w2);

    float e1 = 1.001f * __expf(pbo1[t] * __powf(r / rosi[t], pbo2[t]));
    float e2 = __expf(pbo3[t] * __powf(r / ropi[t], pbo4[t]));
    float e3 = __expf(pbo5[t] * __powf(r / ropp[t], pbo6[t]));
    float bosi = taperf(e1, 0.001f, 0.002f) * (e1 - 0.001f);
    float bopi = taperf(e2, 0.001f, 0.002f) * e2;
    float bopp = taperf(e3, 0.001f, 0.002f) * e3;

    float bo = bosi + bopi + bopp, bpi = bopi + bopp;
    atomicAdd(&Delta[b * NN + i], bo);   atomicAdd(&Delta[b * NN + j], bo);
    atomicAdd(&Dpi[b * NN + i],  bpi);   atomicAdd(&Dpi[b * NN + j],  bpi);
    atomicAdd(&SOa[b * NN + i], bosi);   atomicAdd(&SOa[b * NN + j], bosi);

    // A layout, layer 1: lane = row M, elems 0..2 = K0..K2 (rest zero-padded)
    a1[0] = (_Float16)bosi; a1[1] = (_Float16)bopi; a1[2] = (_Float16)bopp;
  }

  const int m0    = lo ? 0 : 8;   // C rows this lane holds
  const int kbase = lo ? 0 : 8;   // A K-range this lane holds

  // ---- layer 1: h1 = sigmoid(X @ Wi)
  v8f d = wmma16(a1, bw1, czero);
  lds_fence();
  for (int m = 0; m < 8; ++m)
    lds[wave][(m0 + m) * 16 + col] = (_Float16)sigf(d[m]);
  lds_fence();
  v16h a = {};
  for (int e = 0; e < HH; ++e) a[e] = lds[wave][col * 16 + kbase + e];

  // ---- layer 2
  d = wmma16(a, bw2, cb2);
  lds_fence();
  for (int m = 0; m < 8; ++m)
    lds[wave][(m0 + m) * 16 + col] = (_Float16)sigf(d[m]);
  lds_fence();
  a = (v16h){};
  for (int e = 0; e < HH; ++e) a[e] = lds[wave][col * 16 + kbase + e];

  // ---- layer 3
  d = wmma16(a, bw3, cb3);
  lds_fence();
  for (int m = 0; m < 8; ++m)
    lds[wave][(m0 + m) * 16 + col] = (_Float16)sigf(d[m]);
  lds_fence();
  a = (v16h){};
  for (int e = 0; e < HH; ++e) a[e] = lds[wave][col * 16 + kbase + e];

  // ---- layer 4: logits in column 0 only
  d = wmma16(a, bw4, cb4);
  if (col == 0) {  // lane 0: rows 0..7, lane 16: rows 8..15
    float s = 0.0f;
    for (int m = 0; m < 8; ++m) s += -Desi[t] * sigf(d[m]);
    atomicAdd(&out[b], s);
  }
}

// ---------------------------------------------------------------- per-atom energies
// grid = 96 blocks x 256 threads; 2 blocks per batch.
__global__ __launch_bounds__(256)
void atom_kernel(const int* __restrict__ atom_type,
                 const float* __restrict__ sp_val,  const float* __restrict__ sp_vale,
                 const float* __restrict__ sp_lp2,  const float* __restrict__ sp_ov2,
                 const float* __restrict__ sp_ov5,
                 const float* __restrict__ Delta, const float* __restrict__ Dpi,
                 const float* __restrict__ SOa, float* __restrict__ out) {
  __shared__ float red[256];
  const int b = blockIdx.x >> 1;
  const int n = ((blockIdx.x & 1) << 8) + threadIdx.x;

  const int at = atom_type[n];
  const float val = sp_val[at], vale = sp_vale[at], lp2 = sp_lp2[at];
  const float ov2 = sp_ov2[at], ov5 = sp_ov5[at];
  const float D   = Delta[b * NN + n];
  const float dpi = Dpi[b * NN + n];
  const float so  = SOa[b * NN + n];

  const float Nlp = 0.5f * (vale - val);
  const float de  = 0.5f * (D - vale);
  const float De  = fminf(ceilf(de), 0.0f);       // -relu(-ceil(de))
  const float u   = 1.0f + de - De;
  const float nlp = -De + __expf(-40.0f * u * u);  // LP1*4 = 40
  const float Dlp = fmaxf(Nlp - nlp + 1.0f, 0.0f) - 1.0f;
  const float Elone = lp2 * Dlp * __builtin_amdgcn_rcpf(1.0f + __expf(-75.0f * Dlp));

  const float dlpc = D - val -
      Dlp * __builtin_amdgcn_rcpf(1.0f + 5.0f * __expf(3.0f * dpi)); // OVUN3,OVUN4
  const float den  = dlpc + val;
  const float otr  = __builtin_amdgcn_rcpf((den != 0.0f) ? den : 1e-8f);
  const float Eover = so * otr * dlpc * sigf(-ov2 * dlpc);

  const float expeu1 = __expf(6.0f * dlpc);                 // OVUN6
  const float eu1    = sigf(ov2 * dlpc);
  const float eu2    = __builtin_amdgcn_rcpf(1.0f + __expf(12.0f * dpi)); // OVUN7,OVUN8
  const float Eunder = -ov5 * (1.0f - expeu1) * eu1 * eu2;

  red[threadIdx.x] = Elone + Eover + Eunder;
  __syncthreads();
  for (int s = 128; s > 0; s >>= 1) {
    if (threadIdx.x < s) red[threadIdx.x] += red[threadIdx.x + s];
    __syncthreads();
  }
  if (threadIdx.x == 0) atomicAdd(&out[b], red[0]);
}

// ---------------------------------------------------------------- launch
extern "C" void kernel_launch(void* const* d_in, const int* in_sizes, int n_in,
                              void* d_out, int out_size, void* d_ws, size_t ws_size,
                              hipStream_t stream) {
  (void)in_sizes; (void)n_in; (void)out_size; (void)ws_size;
  const float* x     = (const float*)d_in[0];
  const float* cell  = (const float*)d_in[1];
  const float* rcell = (const float*)d_in[2];
  const int*   bdid  = (const int*)d_in[3];
  const int*   atype = (const int*)d_in[4];
  const float* rosi  = (const float*)d_in[5];
  const float* pbo1  = (const float*)d_in[6];
  const float* pbo2  = (const float*)d_in[7];
  const float* ropi  = (const float*)d_in[8];
  const float* pbo3  = (const float*)d_in[9];
  const float* pbo4  = (const float*)d_in[10];
  const float* ropp  = (const float*)d_in[11];
  const float* pbo5  = (const float*)d_in[12];
  const float* pbo6  = (const float*)d_in[13];
  const float* Desi  = (const float*)d_in[14];
  const float* spval = (const float*)d_in[15];
  const float* spvle = (const float*)d_in[16];
  const float* splp2 = (const float*)d_in[17];
  const float* spov2 = (const float*)d_in[18];
  const float* spov5 = (const float*)d_in[19];
  const float* fwi   = (const float*)d_in[20];
  const float* fw    = (const float*)d_in[21];
  const float* fb    = (const float*)d_in[22];
  const float* fwo   = (const float*)d_in[23];
  const float* fbo   = (const float*)d_in[24];

  float* out   = (float*)d_out;
  float* Delta = (float*)d_ws;
  float* Dpi   = Delta + BB * NN;
  float* SOa   = Dpi + BB * NN;
  const int nws = 3 * BB * NN;

  zero_kernel<<<(nws + 255) / 256, 256, 0, stream>>>(out, Delta, nws);
  bond_kernel<<<1536, 256, 0, stream>>>(x, cell, rcell, bdid,
                                        rosi, pbo1, pbo2, ropi, pbo3, pbo4,
                                        ropp, pbo5, pbo6, Desi,
                                        fwi, fw, fb, fwo, fbo,
                                        Delta, Dpi, SOa, out);
  atom_kernel<<<96, 256, 0, stream>>>(atype, spval, spvle, splp2, spov2, spov5,
                                      Delta, Dpi, SOa, out);
}
